// GRU_74397423501869
// MI455X (gfx1250) — compile-verified, matrix operand order checked by reading, not run on hardware
//
#include <hip/hip_runtime.h>
#include <hip/hip_bf16.h>

#define S_LEN 64
#define B_SZ  64
#define HIDN  512
#define EMBD  512
#define VOC   10000

typedef __attribute__((ext_vector_type(16))) __bf16 v16bf;
typedef __attribute__((ext_vector_type(8)))  __bf16 v8bf;
typedef __attribute__((ext_vector_type(8)))  float  v8f;

__device__ __forceinline__ float sigf(float x){ return 1.0f/(1.0f+__expf(-x)); }

// WMMA 16x16x32 bf16 A/B fragment load (row-major source, per-lane K-halves):
// lane<16 : row=lane,    K = [k..k+7]  and [k+16..k+23]
// lane>=16: row=lane-16, K = [k+8..k+15] and [k+24..k+31]
__device__ __forceinline__ v16bf load_frag(const __bf16* p){
  v8bf lo = *(const v8bf*)p;
  v8bf hi = *(const v8bf*)(p + 16);
  return __builtin_shufflevector(lo, hi, 0,1,2,3,4,5,6,7,8,9,10,11,12,13,14,15);
}

// ---------------- prep kernels ----------------
// out[N,K] = (bf16) in[K,N]
__global__ void k_transpose_bf(const float* __restrict__ in, __bf16* __restrict__ out,
                               int K, int N) {
  int idx = blockIdx.x*blockDim.x + threadIdx.x;
  if (idx >= K*N) return;
  int n = idx / K, k = idx - n*K;
  out[idx] = (__bf16)in[k*N + n];
}

__global__ void k_convert_bf(const float* __restrict__ in, __bf16* __restrict__ out, int n) {
  int idx = blockIdx.x*blockDim.x + threadIdx.x;
  if (idx < n) out[idx] = (__bf16)in[idx];
}

__global__ void k_embed(const int* __restrict__ tok, const float* __restrict__ emb,
                        __bf16* __restrict__ X) {
  int idx = blockIdx.x*blockDim.x + threadIdx.x;
  if (idx >= S_LEN*B_SZ*EMBD) return;
  int r = idx / EMBD, c = idx - r*EMBD;
  X[idx] = (__bf16)emb[(size_t)tok[r]*EMBD + c];
}

__global__ void k_init_h(const float* __restrict__ hidden, float* h0f, float* h1f,
                         __bf16* h0b, __bf16* h1b) {
  int idx = blockIdx.x*blockDim.x + threadIdx.x;
  if (idx >= B_SZ*HIDN) return;
  float a = hidden[idx];
  float b = hidden[B_SZ*HIDN + idx];
  h0f[idx]=a; h1f[idx]=b; h0b[idx]=(__bf16)a; h1b[idx]=(__bf16)b;
}

__global__ void k_final_h(const float* __restrict__ h0f, const float* __restrict__ h1f,
                          float* __restrict__ out) {
  int idx = blockIdx.x*blockDim.x + threadIdx.x;
  if (idx >= B_SZ*HIDN) return;
  out[idx] = h0f[idx];
  out[B_SZ*HIDN + idx] = h1f[idx];
}

// ---------------- generic WMMA GEMM: C[M,N] = A[M,K] * BT[N,K] (+bias[N]) ----------------
__global__ void k_gemm_bf16(const __bf16* __restrict__ A, const __bf16* __restrict__ BT,
                            const float* __restrict__ bias, float* __restrict__ C,
                            int M, int N, int K) {
  const int lane = threadIdx.x & 31;
  const int wave = blockIdx.x*(blockDim.x >> 5) + (threadIdx.x >> 5);
  const int ntile = N >> 4;
  const int mt = wave / ntile;
  const int nt = wave - mt*ntile;
  if (mt >= (M >> 4)) return;            // uniform per wave -> EXEC stays all-ones
  const int l15  = lane & 15;
  const int ksel = (lane >> 4) << 3;
  const __bf16* ap = A  + (size_t)((mt<<4) + l15)*K + ksel;
  const __bf16* bp = BT + (size_t)((nt<<4) + l15)*K + ksel;
  v8f acc = {};
  #pragma unroll 4
  for (int k = 0; k < K; k += 32) {
    v16bf a = load_frag(ap + k);
    v16bf b = load_frag(bp + k);
    acc = __builtin_amdgcn_wmma_f32_16x16x32_bf16(false, a, false, b, (short)0, acc, false, false);
  }
  const int col = (nt<<4) + l15;
  const float bv = bias ? bias[col] : 0.0f;
  const int m0 = (mt<<4) + ((lane>>4)<<3);
  #pragma unroll
  for (int r = 0; r < 8; ++r)
    C[(size_t)(m0+r)*N + col] = acc[r] + bv;
}

// ---------------- decoder GEMM: LDS-tiled + register-blocked ----------------
// C[4096,10000] = A[4096,512] * BT[10000,512] + bias.  Block = 128(M) x 80(N),
// grid = 32 x 125 blocks.  B-panel (80x512 bf16, contiguous rows) staged to LDS
// via gfx1250 async-LDS loads; each wave computes 1 M-tile x 5 N-tiles, reusing
// the A fragment 5x and reading B fragments from LDS.
#define DEC_BM 128
#define DEC_BN 80
#define DEC_NB 125                       // VOC / DEC_BN
#define BS_STRIDE 520                    // 512 + 8 elems (16B row pad, keeps 16B align)

__global__ void __launch_bounds__(256)
k_dec_gemm(const __bf16* __restrict__ A, const __bf16* __restrict__ BT,
           const float* __restrict__ bias, float* __restrict__ C) {
  __shared__ __bf16 Bs[DEC_BN * BS_STRIDE];

  const int tid  = threadIdx.x;
  const int lane = tid & 31;
  const int wv   = tid >> 5;             // 0..7
  const int mblk = blockIdx.x / DEC_NB;
  const int nblk = blockIdx.x - mblk*DEC_NB;

  // ---- stage B panel (80 rows x 512 K = 80KB, contiguous) into LDS, async ----
  const __bf16* src = BT + (size_t)nblk * DEC_BN * HIDN;
  #pragma unroll 4
  for (int c = tid; c < (DEC_BN*HIDN*2)/16; c += 256) {   // 5120 16B chunks, 20/thread
    int row = c >> 6;                    // 64 chunks per 1024B row
    int cp  = c & 63;
    unsigned laddr = (unsigned)(uintptr_t)(&Bs[row * BS_STRIDE]) + cp*16;
    unsigned long long gaddr = (unsigned long long)(uintptr_t)(src + (size_t)row*HIDN) + cp*16;
    asm volatile("global_load_async_to_lds_b128 %0, %1, off"
                 :: "v"(laddr), "v"(gaddr) : "memory");
  }
  asm volatile("s_wait_asynccnt 0x0" ::: "memory");
  __syncthreads();

  // ---- compute: each wave owns M rows [mblk*128 + wv*16, +16), all 5 N tiles ----
  const int l15  = lane & 15;
  const int ksel = (lane >> 4) << 3;
  const __bf16* ap  = A + (size_t)(mblk*DEC_BM + wv*16 + l15)*HIDN + ksel;
  const __bf16* bs0 = &Bs[(size_t)l15 * BS_STRIDE + ksel];

  v8f acc[5] = {};
  for (int k = 0; k < HIDN; k += 32) {
    v16bf a = load_frag(ap + k);
    #pragma unroll
    for (int j = 0; j < 5; ++j) {
      v16bf b = load_frag(bs0 + j*16*BS_STRIDE + k);
      acc[j] = __builtin_amdgcn_wmma_f32_16x16x32_bf16(false, a, false, b,
                                                       (short)0, acc[j], false, false);
    }
  }

  const int m0 = mblk*DEC_BM + wv*16 + ((lane>>4)<<3);
  #pragma unroll
  for (int j = 0; j < 5; ++j) {
    const int col = nblk*DEC_BN + j*16 + l15;
    const float bv = bias[col];
    #pragma unroll
    for (int r = 0; r < 8; ++r)
      C[(size_t)(m0+r)*VOC + col] = acc[j][r] + bv;
  }
}

// ---------------- GRU stage 1: UH = Hbf @ UhT ; r & z gates ----------------
__global__ void k_gru_gates1(const __bf16* __restrict__ Hbf, const __bf16* __restrict__ UhT,
                             const float* __restrict__ WX, const float* __restrict__ Hf,
                             __bf16* __restrict__ RHbf, float* __restrict__ Z) {
  const int lane = threadIdx.x & 31;
  const int wave = blockIdx.x*(blockDim.x >> 5) + (threadIdx.x >> 5);
  if (wave >= 4*64) return;              // M=64 -> 4 tiles, N=1024 -> 64 tiles
  const int nt = wave & 63;
  const int mt = wave >> 6;
  const int l15  = lane & 15;
  const int ksel = (lane >> 4) << 3;
  const __bf16* ap = Hbf + ((mt<<4)+l15)*HIDN + ksel;
  const __bf16* bp = UhT + ((nt<<4)+l15)*HIDN + ksel;
  v8f acc = {};
  #pragma unroll 4
  for (int k = 0; k < HIDN; k += 32)
    acc = __builtin_amdgcn_wmma_f32_16x16x32_bf16(false, load_frag(ap+k), false, load_frag(bp+k),
                                                  (short)0, acc, false, false);
  const int n  = (nt<<4) + l15;          // branch below is uniform per wave (tile granularity)
  const int m0 = (mt<<4) + ((lane>>4)<<3);
  if (n < HIDN) {                        // r gate -> rh = sigmoid(wx_r + uh_r) * h
    #pragma unroll
    for (int r = 0; r < 8; ++r) {
      int m = m0 + r;
      float rg = sigf(WX[m*3*HIDN + n] + acc[r]);
      RHbf[m*HIDN + n] = (__bf16)(rg * Hf[m*HIDN + n]);
    }
  } else {                               // z gate
    int n2 = n - HIDN;
    #pragma unroll
    for (int r = 0; r < 8; ++r) {
      int m = m0 + r;
      Z[m*HIDN + n2] = sigf(WX[m*3*HIDN + HIDN + n2] + acc[r]);
    }
  }
}

// ---------------- GRU stage 2: HT = RHbf @ UhtT ; h_new = (1-z)h + z tanh(wx_h + HT) ----------------
__global__ void k_gru_gates2(const __bf16* __restrict__ RHbf, const __bf16* __restrict__ UhtT,
                             const float* __restrict__ WX, const float* __restrict__ Z,
                             float* __restrict__ Hf, __bf16* __restrict__ Hbf,
                             __bf16* __restrict__ H2out) {
  const int lane = threadIdx.x & 31;
  const int wave = blockIdx.x*(blockDim.x >> 5) + (threadIdx.x >> 5);
  if (wave >= 4*32) return;              // M=64 -> 4 tiles, N=512 -> 32 tiles
  const int nt = wave & 31;
  const int mt = wave >> 5;
  const int l15  = lane & 15;
  const int ksel = (lane >> 4) << 3;
  const __bf16* ap = RHbf + ((mt<<4)+l15)*HIDN + ksel;
  const __bf16* bp = UhtT + ((nt<<4)+l15)*HIDN + ksel;
  v8f acc = {};
  #pragma unroll 4
  for (int k = 0; k < HIDN; k += 32)
    acc = __builtin_amdgcn_wmma_f32_16x16x32_bf16(false, load_frag(ap+k), false, load_frag(bp+k),
                                                  (short)0, acc, false, false);
  const int n  = (nt<<4) + l15;
  const int m0 = (mt<<4) + ((lane>>4)<<3);
  #pragma unroll
  for (int r = 0; r < 8; ++r) {
    int m = m0 + r;
    float ht = tanhf(WX[m*3*HIDN + 2*HIDN + n] + acc[r]);
    float z  = Z[m*HIDN + n];
    float hn = (1.0f - z)*Hf[m*HIDN + n] + z*ht;
    Hf[m*HIDN + n] = hn;
    __bf16 hb = (__bf16)hn;
    Hbf[m*HIDN + n] = hb;
    if (H2out) H2out[m*HIDN + n] = hb;
  }
}

extern "C" void kernel_launch(void* const* d_in, const int* in_sizes, int n_in,
                              void* d_out, int out_size, void* d_ws, size_t ws_size,
                              hipStream_t stream) {
  (void)in_sizes; (void)n_in; (void)out_size; (void)ws_size;
  const int*   inputs = (const int*)  d_in[0];
  const float* hidden = (const float*)d_in[1];
  const float* emb    = (const float*)d_in[2];
  const float* Wx     = (const float*)d_in[3];   // (2, 512, 1536)
  const float* Uh     = (const float*)d_in[4];   // (2, 512, 1024)
  const float* Uht    = (const float*)d_in[5];   // (2, 512, 512)
  const float* b_rzh  = (const float*)d_in[6];   // (2, 1536)
  const float* dec_w  = (const float*)d_in[7];   // (10000, 512) — already N x K for WMMA B
  const float* dec_b  = (const float*)d_in[8];   // (10000,)
  float* out = (float*)d_out;

  // -------- workspace layout (256B-aligned slabs) --------
  size_t off = 0;
  char* ws = (char*)d_ws;
  auto alloc = [&](size_t bytes) -> void* {
    void* p = ws + off;
    off += bytes; off = (off + 255) & ~(size_t)255;
    return p;
  };
  __bf16* Xbf   = (__bf16*)alloc((size_t)S_LEN*B_SZ*EMBD*2);       // 4 MB
  float*  WX0   = (float*) alloc((size_t)S_LEN*B_SZ*3*HIDN*4);     // 25 MB
  __bf16* Wx0T  = (__bf16*)alloc((size_t)3*HIDN*EMBD*2);
  __bf16* Wx1T  = (__bf16*)alloc((size_t)3*HIDN*HIDN*2);
  __bf16* Uh0T  = (__bf16*)alloc((size_t)2*HIDN*HIDN*2);
  __bf16* Uh1T  = (__bf16*)alloc((size_t)2*HIDN*HIDN*2);
  __bf16* Uht0T = (__bf16*)alloc((size_t)HIDN*HIDN*2);
  __bf16* Uht1T = (__bf16*)alloc((size_t)HIDN*HIDN*2);
  __bf16* DWbf  = (__bf16*)alloc((size_t)VOC*HIDN*2);              // 10 MB
  float*  h0f   = (float*) alloc((size_t)B_SZ*HIDN*4);
  float*  h1f   = (float*) alloc((size_t)B_SZ*HIDN*4);
  __bf16* h0b   = (__bf16*)alloc((size_t)B_SZ*HIDN*2);
  __bf16* h1b   = (__bf16*)alloc((size_t)B_SZ*HIDN*2);
  __bf16* RHbf  = (__bf16*)alloc((size_t)B_SZ*HIDN*2);
  float*  Zbuf  = (float*) alloc((size_t)B_SZ*HIDN*4);
  float*  WX1   = (float*) alloc((size_t)B_SZ*3*HIDN*4);
  __bf16* H2    = (__bf16*)alloc((size_t)S_LEN*B_SZ*HIDN*2);       // 4 MB

  const int T = 256;                       // 8 waves / block (wave32)
  auto cdiv = [](int a, int b){ return (a + b - 1) / b; };

  // -------- weight prep: transpose to (N,K) + bf16 --------
  k_transpose_bf<<<cdiv(EMBD*3*HIDN,T), T, 0, stream>>>(Wx,               Wx0T, EMBD, 3*HIDN);
  k_transpose_bf<<<cdiv(HIDN*3*HIDN,T), T, 0, stream>>>(Wx + EMBD*3*HIDN, Wx1T, HIDN, 3*HIDN);
  k_transpose_bf<<<cdiv(HIDN*2*HIDN,T), T, 0, stream>>>(Uh,               Uh0T, HIDN, 2*HIDN);
  k_transpose_bf<<<cdiv(HIDN*2*HIDN,T), T, 0, stream>>>(Uh + HIDN*2*HIDN, Uh1T, HIDN, 2*HIDN);
  k_transpose_bf<<<cdiv(HIDN*HIDN,T),   T, 0, stream>>>(Uht,              Uht0T, HIDN, HIDN);
  k_transpose_bf<<<cdiv(HIDN*HIDN,T),   T, 0, stream>>>(Uht + HIDN*HIDN,  Uht1T, HIDN, HIDN);
  k_convert_bf  <<<cdiv(VOC*HIDN,T),    T, 0, stream>>>(dec_w, DWbf, VOC*HIDN);
  k_embed       <<<cdiv(S_LEN*B_SZ*EMBD,T), T, 0, stream>>>(inputs, emb, Xbf);
  k_init_h      <<<cdiv(B_SZ*HIDN,T),   T, 0, stream>>>(hidden, h0f, h1f, h0b, h1b);

  // -------- batched layer-0 input projection for all timesteps: (4096 x 1536) --------
  {
    int tiles = (S_LEN*B_SZ/16) * (3*HIDN/16);
    k_gemm_bf16<<<cdiv(tiles,8), T, 0, stream>>>(Xbf, Wx0T, b_rzh, WX0,
                                                 S_LEN*B_SZ, 3*HIDN, EMBD);
  }

  // -------- sequential recurrence (kernel boundaries = grid-wide sync) --------
  for (int t = 0; t < S_LEN; ++t) {
    const float* wx0t = WX0 + (size_t)t*B_SZ*3*HIDN;
    k_gru_gates1<<<32, T, 0, stream>>>(h0b, Uh0T, wx0t, h0f, RHbf, Zbuf);
    k_gru_gates2<<<16, T, 0, stream>>>(RHbf, Uht0T, wx0t, Zbuf, h0f, h0b, (__bf16*)nullptr);
    {
      int tiles = (B_SZ/16) * (3*HIDN/16);
      k_gemm_bf16<<<cdiv(tiles,8), T, 0, stream>>>(h0b, Wx1T, b_rzh + 3*HIDN, WX1,
                                                   B_SZ, 3*HIDN, HIDN);
    }
    k_gru_gates1<<<32, T, 0, stream>>>(h1b, Uh1T, WX1, h1f, RHbf, Zbuf);
    k_gru_gates2<<<16, T, 0, stream>>>(RHbf, Uht1T, WX1, Zbuf, h1f, h1b,
                                       H2 + (size_t)t*B_SZ*HIDN);
  }

  // -------- decoder: LDS-tiled (4096 x 10000), async B staging, 5x A reuse --------
  k_dec_gemm<<<(S_LEN*B_SZ/DEC_BM) * DEC_NB, T, 0, stream>>>(H2, DWbf, dec_b, out);

  // -------- h_final (2, 64, 512) appended after logits --------
  k_final_h<<<cdiv(B_SZ*HIDN,T), T, 0, stream>>>(h0f, h1f, out + (size_t)S_LEN*B_SZ*VOC);
}
